// CrossAttentionBlock_58995670777862
// MI455X (gfx1250) — matrix operand order repaired and checked by reading, not compiled, not run on hardware
//
#include <hip/hip_runtime.h>

// ---------------- types ----------------
typedef __bf16 bf16_t;
typedef __attribute__((ext_vector_type(16))) __bf16 v16bf;
typedef __attribute__((ext_vector_type(8)))  float  v8f;

// ---------------- problem constants ----------------
#define BATCH   4
#define FRAMES  16
#define CDIM    512
#define HW      256
#define VLEN    4096      // FRAMES * HW
#define LAUD    1024
#define T_ALL   5120      // VLEN + LAUD
#define HEADS   8
#define CH      64        // CDIM / HEADS
#define ALPF    64        // LAUD / FRAMES
#define O3      1536      // 3 * CDIM
#define NGROUPS 32
#define CPG     16        // CDIM / NGROUPS
#define SCALE2  0.125f    // (1/sqrt(sqrt(64)))^2
#define EPSV    1e-5f

// ---------------- helpers ----------------
__device__ __forceinline__ float warp_sum(float v) {
#pragma unroll
  for (int m = 16; m >= 1; m >>= 1) v += __shfl_xor(v, m, 32);
  return v;
}

// A-matrix bf16 fragment (16x32 MxK): lane holds M=lane&15.
// Elements = two contiguous 8-element runs of the lane's 32-long K-run,
// at element offsets 8*half and 8*half+16  (ISA 7.12.2 16-bit A layout).
__device__ __forceinline__ v16bf load_a_frag(const bf16_t* kbase, int hf) {
  union { v16bf v; uint4 u[2]; } t;
  t.u[0] = *(const uint4*)(kbase + 8 * hf);
  t.u[1] = *(const uint4*)(kbase + 8 * hf + 16);
  return t.v;
}

// B-matrix bf16 fragment (32x16 KxN): lane holds N=lane&15; elements are two
// contiguous 8-element K-runs at offsets 8*half and 8*half+16 (matches A).
__device__ __forceinline__ v16bf load_b_frag(const bf16_t* kbase, int hf) {
  union { v16bf v; uint4 u[2]; } t;
  t.u[0] = *(const uint4*)(kbase + 8 * hf);
  t.u[1] = *(const uint4*)(kbase + 8 * hf + 16);
  return t.v;
}

__device__ __forceinline__ v8f wmma_bf16(v16bf a, v16bf b, v8f c) {
  return __builtin_amdgcn_wmma_f32_16x16x32_bf16(false, a, false, b,
                                                 (short)0, c, false, false);
}

// CDNA5 async global->LDS copy (VGLOBAL GLOBAL_LOAD_ASYNC_TO_LDS_B128,
// tracked with ASYNCcnt). LDS address = low 32 bits of flat shared pointer.
__device__ __forceinline__ void async_load_b128(void* lds_dst, const void* gsrc) {
  unsigned lds = (unsigned)(size_t)lds_dst;
  unsigned long long ga = (unsigned long long)(size_t)gsrc;
  asm volatile("global_load_async_to_lds_b128 %0, %1, off"
               :: "v"(lds), "v"(ga) : "memory");
}
__device__ __forceinline__ void wait_async() {
  asm volatile("s_wait_asynccnt 0" ::: "memory");
}

// ---------------- GroupNorm (video) -> token-major bf16 [b][t][c] ----------------
__global__ void __launch_bounds__(256)
gn_video(const float* __restrict__ video, const float* __restrict__ scale,
         const float* __restrict__ bias, bf16_t* __restrict__ vt) {
  int b = blockIdx.x >> 5;
  int g = blockIdx.x & 31;
  int tid = threadIdx.x;
  float s = 0.f, s2 = 0.f;
  for (int i = tid; i < CPG * VLEN; i += 256) {
    int j = i >> 12, t = i & 4095;          // read-coalesced over hw
    int c = g * CPG + j;
    int f = t >> 8, hw = t & 255;
    float v = video[(((size_t)(b * FRAMES + f) * CDIM + c) << 8) + hw];
    s += v; s2 += v * v;
  }
  __shared__ float rs[8], rq[8];
  s = warp_sum(s); s2 = warp_sum(s2);
  int wid = tid >> 5, ln = tid & 31;
  if (ln == 0) { rs[wid] = s; rq[wid] = s2; }
  __syncthreads();
  float ts = 0.f, tq = 0.f;
#pragma unroll
  for (int i = 0; i < 8; ++i) { ts += rs[i]; tq += rq[i]; }
  const float inv_n = 1.0f / (float)(CPG * VLEN);
  float mu  = ts * inv_n;
  float var = tq * inv_n - mu * mu;
  float rinv = rsqrtf(var + EPSV);
  for (int i = tid; i < CPG * VLEN; i += 256) {
    int t = i >> 4, j = i & 15;             // write-coalesced over c
    int c = g * CPG + j;
    int f = t >> 8, hw = t & 255;
    float v = video[(((size_t)(b * FRAMES + f) * CDIM + c) << 8) + hw];
    float y = (v - mu) * rinv * scale[c] + bias[c];
    vt[((size_t)b * VLEN + t) * CDIM + c] = (bf16_t)y;
  }
}

// ---------------- GroupNorm (audio) -> token-major bf16 [b][t][c] ----------------
__global__ void __launch_bounds__(256)
gn_audio(const float* __restrict__ audio, const float* __restrict__ scale,
         const float* __restrict__ bias, bf16_t* __restrict__ at) {
  int b = blockIdx.x >> 5;
  int g = blockIdx.x & 31;
  int tid = threadIdx.x;
  float s = 0.f, s2 = 0.f;
  for (int i = tid; i < CPG * LAUD; i += 256) {
    int j = i >> 10, t = i & 1023;
    int c = g * CPG + j;
    float v = audio[(size_t)(b * CDIM + c) * LAUD + t];
    s += v; s2 += v * v;
  }
  __shared__ float rs[8], rq[8];
  s = warp_sum(s); s2 = warp_sum(s2);
  int wid = tid >> 5, ln = tid & 31;
  if (ln == 0) { rs[wid] = s; rq[wid] = s2; }
  __syncthreads();
  float ts = 0.f, tq = 0.f;
#pragma unroll
  for (int i = 0; i < 8; ++i) { ts += rs[i]; tq += rq[i]; }
  const float inv_n = 1.0f / (float)(CPG * LAUD);
  float mu  = ts * inv_n;
  float var = tq * inv_n - mu * mu;
  float rinv = rsqrtf(var + EPSV);
  for (int i = tid; i < CPG * LAUD; i += 256) {
    int t = i >> 4, j = i & 15;
    int c = g * CPG + j;
    float v = audio[(size_t)(b * CDIM + c) * LAUD + t];
    float y = (v - mu) * rinv * scale[c] + bias[c];
    at[((size_t)b * LAUD + t) * CDIM + c] = (bf16_t)y;
  }
}

// ---------------- weight f32 -> bf16 ----------------
__global__ void convert_w(const float* __restrict__ wvf, const float* __restrict__ waf,
                          bf16_t* __restrict__ wv, bf16_t* __restrict__ wa, int n) {
  int i = blockIdx.x * blockDim.x + threadIdx.x;
  if (i < n)            wv[i]     = (bf16_t)wvf[i];
  else if (i < 2 * n)   wa[i - n] = (bf16_t)waf[i - n];
}

// ---------------- QKV GEMM ----------------
// out[o,t] = W[o,:] . X[t,:] + bias[o]   (X token-major)
// block: 256 threads (8 waves), output tile 128(o) x 64(t), K-steps of 32.
// Staging via CDNA5 async global->LDS transfers (ASYNCcnt).
// q,k results -> token-major [b][T_ALL][c]; v -> channel-major [b][c][T_ALL].
__global__ void __launch_bounds__(256)
qkv_gemm(const bf16_t* __restrict__ Wt, const bf16_t* __restrict__ X,
         const float* __restrict__ bias,
         bf16_t* __restrict__ qo, bf16_t* __restrict__ ko, bf16_t* __restrict__ vo,
         int T, int t_off) {
  __shared__ __align__(16) bf16_t ldsA[128 * 32];   // [o][k]
  __shared__ __align__(16) bf16_t ldsB[64 * 32];    // [t][k]
  int b      = blockIdx.z;
  int o_base = blockIdx.y * 128;
  int t_base = blockIdx.x * 64;
  int tid = threadIdx.x;
  int wid = tid >> 5, lane = tid & 31;
  int Ml = lane & 15, hf = lane >> 4;
  const bf16_t* Xb = X + (size_t)b * T * CDIM;

  // per-lane bias for this wave's 8 consecutive o rows
  float bv[8];
  {
    const float4 b0 = *(const float4*)(bias + o_base + wid * 16 + 8 * hf);
    const float4 b1 = *(const float4*)(bias + o_base + wid * 16 + 8 * hf + 4);
    bv[0] = b0.x; bv[1] = b0.y; bv[2] = b0.z; bv[3] = b0.w;
    bv[4] = b1.x; bv[5] = b1.y; bv[6] = b1.z; bv[7] = b1.w;
  }

  const v8f vzero = {0.f, 0.f, 0.f, 0.f, 0.f, 0.f, 0.f, 0.f};
  v8f acc[4];
#pragma unroll
  for (int i = 0; i < 4; ++i) acc[i] = vzero;

  for (int k0 = 0; k0 < CDIM; k0 += 32) {
    __syncthreads();
    {   // A tile: 128 rows(o) x 32(k) => 2 async b128 per thread
      int r = tid >> 1, seg = tid & 1;
      const bf16_t* g = Wt + (size_t)(o_base + r) * CDIM + k0 + seg * 16;
      bf16_t* d = ldsA + r * 32 + seg * 16;
      async_load_b128(d, g);
      async_load_b128(d + 8, g + 8);
    }
    {   // B tile: 64 rows(t) x 32(k) => 1 async b128 per thread
      int r = tid >> 2, seg = tid & 3;
      async_load_b128(ldsB + r * 32 + seg * 8,
                      Xb + (size_t)(t_base + r) * CDIM + k0 + seg * 8);
    }
    wait_async();
    __syncthreads();

    v16bf afrag = load_a_frag(ldsA + (wid * 16 + Ml) * 32, hf);
#pragma unroll
    for (int nt = 0; nt < 4; ++nt) {
      v16bf bfrag = load_b_frag(ldsB + (nt * 16 + Ml) * 32, hf);
      acc[nt] = wmma_bf16(afrag, bfrag, acc[nt]);
    }
  }

  // epilogue: acc element p -> row o0+p (8 consecutive channels) at col t=Ml
  int o0 = o_base + wid * 16 + 8 * hf;
#pragma unroll
  for (int nt = 0; nt < 4; ++nt) {
    int t = t_off + t_base + nt * 16 + Ml;
    union { uint4 u; bf16_t e[8]; } pk;
#pragma unroll
    for (int p = 0; p < 8; ++p) pk.e[p] = (bf16_t)(acc[nt][p] + bv[p]);
    if (o0 < CDIM) {                                  // q: [b][t][c], contiguous
      *(uint4*)(qo + ((size_t)b * T_ALL + t) * CDIM + o0) = pk.u;
    } else if (o0 < 2 * CDIM) {                       // k: [b][t][c], contiguous
      *(uint4*)(ko + ((size_t)b * T_ALL + t) * CDIM + (o0 - CDIM)) = pk.u;
    } else {                                          // v: [b][c][t], strided
#pragma unroll
      for (int p = 0; p < 8; ++p)
        vo[(size_t)(b * CDIM + (o0 - 2 * CDIM + p)) * T_ALL + t] = pk.e[p];
    }
  }
}

// ---------------- video attention: one block (256 thr, 8 waves) per (b,h,f) ----------------
// Q: 256(t) x 64(c) token-major; K: 64(s) x 64(c) token-major; V: 64(c) x 64(s) channel-major.
__global__ void __launch_bounds__(256)
attn_video(const bf16_t* __restrict__ qg, const bf16_t* __restrict__ kg,
           const bf16_t* __restrict__ vg, const float* __restrict__ video,
           float* __restrict__ out_video) {
  __shared__ __align__(16) bf16_t ldsP[256 * 64];   // probs, 32 KB
  int bid = blockIdx.x;
  int b = bid >> 7, h = (bid >> 4) & 7, f = bid & 15;
  int tid = threadIdx.x;
  int wid = tid >> 5, lane = tid & 31;
  int Ml = lane & 15, hf = lane >> 4;

  const bf16_t* q_tok = qg + ((size_t)b * T_ALL + f * HW) * CDIM + h * CH;          // +t*CDIM
  const bf16_t* k_tok = kg + ((size_t)b * T_ALL + VLEN + f * ALPF) * CDIM + h * CH; // +s*CDIM
  const bf16_t* v_ch  = vg + (size_t)(b * CDIM + h * CH) * T_ALL + VLEN + f * ALPF; // +c*T_ALL

  const v8f vzero = {0.f, 0.f, 0.f, 0.f, 0.f, 0.f, 0.f, 0.f};
  v8f accS[2][4];
#pragma unroll
  for (int i = 0; i < 2; ++i)
#pragma unroll
    for (int j = 0; j < 4; ++j) accS[i][j] = vzero;

  // stage 1: S = Q K^T  (each wave: M-tiles {2w,2w+1} x 4 N-tiles of s)
#pragma unroll
  for (int ks = 0; ks < 2; ++ks) {
    v16bf bfrag[4];
#pragma unroll
    for (int nt = 0; nt < 4; ++nt)
      bfrag[nt] = load_b_frag(k_tok + (size_t)(nt * 16 + Ml) * CDIM + ks * 32, hf);
#pragma unroll
    for (int mi = 0; mi < 2; ++mi) {
      int m = wid * 2 + mi;
      v16bf afrag = load_a_frag(q_tok + (size_t)(m * 16 + Ml) * CDIM + ks * 32, hf);
#pragma unroll
      for (int nt = 0; nt < 4; ++nt)
        accS[mi][nt] = wmma_bf16(afrag, bfrag[nt], accS[mi][nt]);
    }
  }

  // softmax over s (64); store P (bf16) to this wave's private LDS rows
#pragma unroll
  for (int mi = 0; mi < 2; ++mi) {
    int m = wid * 2 + mi;
#pragma unroll
    for (int p = 0; p < 8; ++p) {
      float xs[4];
      float mx = -3.0e38f;
#pragma unroll
      for (int nt = 0; nt < 4; ++nt) { xs[nt] = accS[mi][nt][p] * SCALE2; mx = fmaxf(mx, xs[nt]); }
#pragma unroll
      for (int d = 8; d >= 1; d >>= 1) mx = fmaxf(mx, __shfl_xor(mx, d, 16));
      float sum = 0.f;
#pragma unroll
      for (int nt = 0; nt < 4; ++nt) { xs[nt] = __expf(xs[nt] - mx); sum += xs[nt]; }
#pragma unroll
      for (int d = 8; d >= 1; d >>= 1) sum += __shfl_xor(sum, d, 16);
      float rden = 1.0f / sum;
      int t_loc = m * 16 + p + 8 * hf;
#pragma unroll
      for (int nt = 0; nt < 4; ++nt)
        ldsP[t_loc * 64 + nt * 16 + Ml] = (bf16_t)(xs[nt] * rden);
    }
  }

  // stage 2: O = P V^T   (O[t,c'] = sum_s P[t,s] V[c',s])
  v8f accO[2][4];
#pragma unroll
  for (int i = 0; i < 2; ++i)
#pragma unroll
    for (int j = 0; j < 4; ++j) accO[i][j] = vzero;

#pragma unroll
  for (int ks = 0; ks < 2; ++ks) {
    v16bf bfrag[4];
#pragma unroll
    for (int nt = 0; nt < 4; ++nt)
      bfrag[nt] = load_b_frag(v_ch + (size_t)(nt * 16 + Ml) * T_ALL + ks * 32, hf);
#pragma unroll
    for (int mi = 0; mi < 2; ++mi) {
      int m = wid * 2 + mi;
      v16bf afrag = load_a_frag(ldsP + (m * 16 + Ml) * 64 + ks * 32, hf);
#pragma unroll
      for (int nt = 0; nt < 4; ++nt)
        accO[mi][nt] = wmma_bf16(afrag, bfrag[nt], accO[mi][nt]);
    }
  }

  // residual + write video_h: acc element p -> 8 consecutive hw positions
#pragma unroll
  for (int mi = 0; mi < 2; ++mi) {
#pragma unroll
    for (int nt = 0; nt < 4; ++nt) {
      int t0 = (wid * 2 + mi) * 16 + 8 * hf;
      int c = h * CH + nt * 16 + Ml;
      size_t idx = (((size_t)(b * FRAMES + f) * CDIM + c) << 8) + t0;
      float4 r0 = *(const float4*)(video + idx);
      float4 r1 = *(const float4*)(video + idx + 4);
      r0.x += accO[mi][nt][0]; r0.y += accO[mi][nt][1];
      r0.z += accO[mi][nt][2]; r0.w += accO[mi][nt][3];
      r1.x += accO[mi][nt][4]; r1.y += accO[mi][nt][5];
      r1.z += accO[mi][nt][6]; r1.w += accO[mi][nt][7];
      *(float4*)(out_video + idx)     = r0;
      *(float4*)(out_video + idx + 4) = r1;
    }
  }
}

// ---------------- audio attention: one block (128 thr, 4 waves) per (b,h,f) ----------------
// Q: 64(t) x 64(c) token-major; K: 256(s) x 64(c) token-major; V: 64(c) x 256(s) channel-major.
__global__ void __launch_bounds__(128)
attn_audio(const bf16_t* __restrict__ qg, const bf16_t* __restrict__ kg,
           const bf16_t* __restrict__ vg, const float* __restrict__ audio,
           float* __restrict__ out_audio) {
  __shared__ __align__(16) bf16_t ldsP[64 * 256];   // probs, 32 KB
  int bid = blockIdx.x;
  int b = bid >> 7, h = (bid >> 4) & 7, f = bid & 15;
  int tid = threadIdx.x;
  int wid = tid >> 5, lane = tid & 31;
  int Ml = lane & 15, hf = lane >> 4;

  const bf16_t* q_tok = qg + ((size_t)b * T_ALL + VLEN + f * ALPF) * CDIM + h * CH; // +t*CDIM
  const bf16_t* k_tok = kg + ((size_t)b * T_ALL + f * HW) * CDIM + h * CH;          // +s*CDIM
  const bf16_t* v_ch  = vg + (size_t)(b * CDIM + h * CH) * T_ALL + f * HW;          // +c*T_ALL

  const v8f vzero = {0.f, 0.f, 0.f, 0.f, 0.f, 0.f, 0.f, 0.f};
  v8f accS[16];
#pragma unroll
  for (int i = 0; i < 16; ++i) accS[i] = vzero;

  // stage 1: S = Q K^T  (wave wid owns M-tile wid; 16 N-tiles over s=256)
#pragma unroll
  for (int ks = 0; ks < 2; ++ks) {
    v16bf afrag = load_a_frag(q_tok + (size_t)(wid * 16 + Ml) * CDIM + ks * 32, hf);
#pragma unroll
    for (int nt = 0; nt < 16; ++nt) {
      v16bf bfrag = load_b_frag(k_tok + (size_t)(nt * 16 + Ml) * CDIM + ks * 32, hf);
      accS[nt] = wmma_bf16(afrag, bfrag, accS[nt]);
    }
  }

  // softmax over s (256); store P to this wave's private LDS rows
#pragma unroll
  for (int p = 0; p < 8; ++p) {
    float xs[16];
    float mx = -3.0e38f;
#pragma unroll
    for (int nt = 0; nt < 16; ++nt) { xs[nt] = accS[nt][p] * SCALE2; mx = fmaxf(mx, xs[nt]); }
#pragma unroll
    for (int d = 8; d >= 1; d >>= 1) mx = fmaxf(mx, __shfl_xor(mx, d, 16));
    float sum = 0.f;
#pragma unroll
    for (int nt = 0; nt < 16; ++nt) { xs[nt] = __expf(xs[nt] - mx); sum += xs[nt]; }
#pragma unroll
    for (int d = 8; d >= 1; d >>= 1) sum += __shfl_xor(sum, d, 16);
    float rden = 1.0f / sum;
    int t_loc = wid * 16 + p + 8 * hf;
#pragma unroll
    for (int nt = 0; nt < 16; ++nt)
      ldsP[t_loc * 256 + nt * 16 + Ml] = (bf16_t)(xs[nt] * rden);
  }

  // stage 2: O = P V^T  (K = s = 256 -> 8 k-steps)
  v8f accO[4];
#pragma unroll
  for (int i = 0; i < 4; ++i) accO[i] = vzero;

#pragma unroll
  for (int ks = 0; ks < 8; ++ks) {
    v16bf afrag = load_a_frag(ldsP + (wid * 16 + Ml) * 256 + ks * 32, hf);
#pragma unroll
    for (int nt = 0; nt < 4; ++nt) {
      v16bf bfrag = load_b_frag(v_ch + (size_t)(nt * 16 + Ml) * T_ALL + ks * 32, hf);
      accO[nt] = wmma_bf16(afrag, bfrag, accO[nt]);
    }
  }

  // residual + write audio_h: acc element p -> 8 consecutive l positions
#pragma unroll
  for (int nt = 0; nt < 4; ++nt) {
    int t0 = wid * 16 + 8 * hf;
    int c = h * CH + nt * 16 + Ml;
    size_t idx = (size_t)(b * CDIM + c) * LAUD + f * ALPF + t0;
    float4 r0 = *(const float4*)(audio + idx);
    float4 r1 = *(const float4*)(audio + idx + 4);
    r0.x += accO[nt][0]; r0.y += accO[nt][1];
    r0.z += accO[nt][2]; r0.w += accO[nt][3];
    r1.x += accO[nt][4]; r1.y += accO[nt][5];
    r1.z += accO[nt][6]; r1.w += accO[nt][7];
    *(float4*)(out_audio + idx)     = r0;
    *(float4*)(out_audio + idx + 4) = r1;
  }
}

// ---------------- launch ----------------
extern "C" void kernel_launch(void* const* d_in, const int* in_sizes, int n_in,
                              void* d_out, int out_size, void* d_ws, size_t ws_size,
                              hipStream_t stream) {
  (void)in_sizes; (void)n_in; (void)out_size; (void)ws_size;
  const float* audio   = (const float*)d_in[0];
  // d_in[1] = time_step (unused by the math)
  const float* video   = (const float*)d_in[2];
  const float* v_scale = (const float*)d_in[3];
  const float* v_bias  = (const float*)d_in[4];
  const float* a_scale = (const float*)d_in[5];
  const float* a_bias  = (const float*)d_in[6];
  const float* v_qkv_w = (const float*)d_in[7];
  const float* v_qkv_b = (const float*)d_in[8];
  const float* a_qkv_w = (const float*)d_in[9];
  const float* a_qkv_b = (const float*)d_in[10];

  float* out_video = (float*)d_out;
  float* out_audio = out_video + (size_t)BATCH * FRAMES * CDIM * HW;

  // workspace carve-up (bf16): ~87 MB total
  bf16_t* vt = (bf16_t*)d_ws;                          // [b][t][c] B*VLEN*C
  bf16_t* at = vt + (size_t)BATCH * VLEN * CDIM;       // [b][t][c] B*L*C
  bf16_t* wv = at + (size_t)BATCH * LAUD * CDIM;       // [o][c]
  bf16_t* wa = wv + (size_t)O3 * CDIM;                 // [o][c]
  bf16_t* qb = wa + (size_t)O3 * CDIM;                 // [b][t][c] B*T_ALL*C
  bf16_t* kb = qb + (size_t)BATCH * T_ALL * CDIM;      // [b][t][c]
  bf16_t* vb = kb + (size_t)BATCH * T_ALL * CDIM;      // [b][c][t]

  gn_video<<<dim3(BATCH * NGROUPS), 256, 0, stream>>>(video, v_scale, v_bias, vt);
  gn_audio<<<dim3(BATCH * NGROUPS), 256, 0, stream>>>(audio, a_scale, a_bias, at);

  int nW = O3 * CDIM;
  convert_w<<<(2 * nW + 255) / 256, 256, 0, stream>>>(v_qkv_w, a_qkv_w, wv, wa, nW);

  qkv_gemm<<<dim3(VLEN / 64, O3 / 128, BATCH), 256, 0, stream>>>(
      wv, vt, v_qkv_b, qb, kb, vb, VLEN, 0);
  qkv_gemm<<<dim3(LAUD / 64, O3 / 128, BATCH), 256, 0, stream>>>(
      wa, at, a_qkv_b, qb, kb, vb, LAUD, VLEN);

  attn_video<<<dim3(BATCH * HEADS * FRAMES), 256, 0, stream>>>(
      qb, kb, vb, video, out_video);
  attn_audio<<<dim3(BATCH * HEADS * FRAMES), 128, 0, stream>>>(
      qb, kb, vb, audio, out_audio);
}
